// Attention_8589934592227
// MI455X (gfx1250) — matrix operand order repaired and checked by reading, not compiled, not run on hardware
//
#include <hip/hip_runtime.h>
#include <hip/hip_bf16.h>
#include <stdint.h>

// Problem constants (from reference): B=4, S=2048, D=1024, H=16, HD=64
#define Bq   4
#define Sq   2048
#define Dq   1024
#define Hq   16
#define HDq  64

typedef __attribute__((ext_vector_type(16))) __bf16 bf16x16;
typedef __attribute__((ext_vector_type(8)))  float  floatx8;
typedef int v4i __attribute__((vector_size(16)));   // v4i32, matches builtin param

#define GLOBAL_AS __attribute__((address_space(1)))
#define LDS_AS    __attribute__((address_space(3)))

// ---- CDNA5 async global->LDS path (guarded; sync fallback keeps code correct) ----
#if defined(__has_builtin)
#  if __has_builtin(__builtin_amdgcn_global_load_async_to_lds_b128)
#    define HAVE_ASYNC_LDS 1
#  endif
#endif
#ifndef HAVE_ASYNC_LDS
#  define HAVE_ASYNC_LDS 0
#endif

#if HAVE_ASYNC_LDS
#  if __has_builtin(__builtin_amdgcn_s_wait_asynccnt)
#    define WAIT_ASYNC(n) __builtin_amdgcn_s_wait_asynccnt(n)
#  else
#    define WAIT_ASYNC(n) asm volatile("s_wait_asynccnt %0" :: "i"(n) : "memory")
#  endif
__device__ __forceinline__ void async_copy_b128(const unsigned short* g, unsigned short* l) {
  __builtin_amdgcn_global_load_async_to_lds_b128(
      (GLOBAL_AS v4i*)g, (LDS_AS v4i*)l, 0, 0);
}
#else
#  define WAIT_ASYNC(n) ((void)0)
#endif

// ---------- bf16 helpers (RNE) ----------
__device__ __forceinline__ unsigned short f2bf_bits(float f) {
  unsigned int u = __float_as_uint(f);
  u += 0x7FFFu + ((u >> 16) & 1u);
  return (unsigned short)(u >> 16);
}
__device__ __forceinline__ __bf16 bits2bf(unsigned short s) {
  return __builtin_bit_cast(__bf16, s);
}

union FragU { uint4 q[2]; bf16x16 v; };

// A fragment: 16x32 (MxK) bf16, row-major source [m][k], stride ld elements.
// Per-lane data is two contiguous 16B chunks -> 2x b128 loads.
__device__ __forceinline__ bf16x16 load_frag_a_fast(const unsigned short* src, int ld) {
  int lane = threadIdx.x & 31;
  int half = lane >> 4, m = lane & 15;
  const unsigned short* base = src + (size_t)m * ld + half * 8;
  FragU f;
  f.q[0] = *reinterpret_cast<const uint4*>(base);        // k = half*8 + 0..7
  f.q[1] = *reinterpret_cast<const uint4*>(base + 16);   // k = 16 + half*8 + 0..7
  return f.v;
}

// B fragment: 32x16 (KxN) bf16, from TRANSPOSED source [n][k], stride ld elements.
__device__ __forceinline__ bf16x16 load_frag_b_fastT(const unsigned short* srcT, int ld) {
  int lane = threadIdx.x & 31;
  int n = lane & 15, kb = (lane >> 4) * 8;
  const unsigned short* base = srcT + (size_t)n * ld + kb;
  FragU f;
  f.q[0] = *reinterpret_cast<const uint4*>(base);        // k = kb + 0..7
  f.q[1] = *reinterpret_cast<const uint4*>(base + 16);   // k = 16 + kb + 0..7
  return f.v;
}

// ---------- fp32 -> bf16 conversion ----------
__global__ void f32_to_bf16_kernel(const float* __restrict__ in,
                                   unsigned short* __restrict__ out, size_t n) {
  size_t i = (size_t)blockIdx.x * blockDim.x + threadIdx.x;
  if (i < n) out[i] = f2bf_bits(in[i]);
}

// ---------- GEMM staging helpers ----------
struct BTile { uint4 r0, r1; };

__device__ __forceinline__ BTile load_btile(const unsigned short* Bm, int N,
                                            int k0, int bn, int tid) {
  int kp = tid & 15, nb = (tid >> 4) * 8;
  const unsigned short* g0 = Bm + (size_t)(k0 + 2 * kp) * N + bn * 64 + nb;
  BTile t;
  t.r0 = *reinterpret_cast<const uint4*>(g0);
  t.r1 = *reinterpret_cast<const uint4*>(g0 + N);
  return t;
}

__device__ __forceinline__ void store_btileT(unsigned short* BsT, BTile t, int tid) {
  int kp = tid & 15, nb = (tid >> 4) * 8;
  unsigned int w0[4] = {t.r0.x, t.r0.y, t.r0.z, t.r0.w};
  unsigned int w1[4] = {t.r1.x, t.r1.y, t.r1.z, t.r1.w};
  unsigned int* B32 = reinterpret_cast<unsigned int*>(BsT);
#pragma unroll
  for (int i = 0; i < 8; ++i) {
    unsigned int lo = (i & 1) ? (w0[i >> 1] >> 16) : (w0[i >> 1] & 0xffffu);
    unsigned int hi = (i & 1) ? (w1[i >> 1] >> 16) : (w1[i >> 1] & 0xffffu);
    B32[(nb + i) * 16 + kp] = lo | (hi << 16);
  }
}

// A tile 128x32, one row per thread (wave-private rows: wave w stages & uses rows 32w..32w+31)
__device__ __forceinline__ void stage_a(const unsigned short* A, unsigned short* AsBuf,
                                        int K, int bm, int k0, int tid) {
  const unsigned short* g = A + (size_t)(bm * 128 + tid) * K + k0;
  unsigned short* l = &AsBuf[tid * 32];
#if HAVE_ASYNC_LDS
#pragma unroll
  for (int i = 0; i < 4; ++i) async_copy_b128(g + i * 8, l + i * 8);
#else
  const uint4* gq = reinterpret_cast<const uint4*>(g);
  uint4* lq = reinterpret_cast<uint4*>(l);
  lq[0] = gq[0]; lq[1] = gq[1]; lq[2] = gq[2]; lq[3] = gq[3];
#endif
}

// ---------- Tiled WMMA GEMM: C[M,N] = A[M,K](bf16)*B[K,N](bf16) + bias ----------
// Block tile 128x64, 128 threads (4 waves), double-buffered LDS, async A staging.
// MODE 0: out fp32 row-major [M,N]; MODE 1: bf16 in [B,H,S,HD] layout, *scale.
template <int MODE>
__global__ __launch_bounds__(128) void gemm_bf16_wmma(
    const unsigned short* __restrict__ A, const unsigned short* __restrict__ Bm,
    const float* __restrict__ bias, void* __restrict__ out,
    int N, int K, float scale) {
  __shared__ unsigned short As[2][128 * 32];   // [m][k]
  __shared__ unsigned short BsT[2][64 * 32];   // [n][k] (transposed B tile)

  const int bm = blockIdx.x, bn = blockIdx.y;
  const int tid = threadIdx.x, wave = tid >> 5, lane = tid & 31;

  floatx8 acc[2][4] = {};

  // prologue: tile 0 in flight
  stage_a(A, As[0], K, bm, 0, tid);
  BTile bt = load_btile(Bm, N, 0, bn, tid);

  const int T = K / 32;
  for (int t = 0; t < T; ++t) {
    const int cur = t & 1;
    store_btileT(BsT[cur], bt, tid);
    if (t + 1 < T) {
      stage_a(A, As[cur ^ 1], K, bm, (t + 1) * 32, tid);   // async, overlaps compute(t)
      bt = load_btile(Bm, N, (t + 1) * 32, bn, tid);
    }
    __syncthreads();                       // BsT[cur] visible; prior readers done
    if (t + 1 < T) { WAIT_ASYNC(4); }      // tile t's 4 asyncs done; t+1's in flight
    else          { WAIT_ASYNC(0); }

    bf16x16 bfr[4];
#pragma unroll
    for (int ns = 0; ns < 4; ++ns)
      bfr[ns] = load_frag_b_fastT(&BsT[cur][(ns * 16) * 32], 32);
#pragma unroll
    for (int mi = 0; mi < 2; ++mi) {
      bf16x16 a = load_frag_a_fast(&As[cur][(wave * 32 + mi * 16) * 32], 32);
#pragma unroll
      for (int ns = 0; ns < 4; ++ns)
        acc[mi][ns] = __builtin_amdgcn_wmma_f32_16x16x32_bf16(
            false, a, false, bfr[ns], (short)0, acc[mi][ns], false, false);
    }
  }

  const int nloc = lane & 15, halfsel = lane >> 4;
#pragma unroll
  for (int mi = 0; mi < 2; ++mi) {
    int rowbase = bm * 128 + wave * 32 + mi * 16 + halfsel * 8;
#pragma unroll
    for (int ns = 0; ns < 4; ++ns) {
      int col = bn * 64 + ns * 16 + nloc;
      float bvv = bias[col];
#pragma unroll
      for (int r = 0; r < 8; ++r) {
        int row = rowbase + r;
        float v = (acc[mi][ns][r] + bvv) * scale;
        if constexpr (MODE == 0) {
          reinterpret_cast<float*>(out)[(size_t)row * N + col] = v;
        } else {
          int b = row >> 11, s = row & (Sq - 1);
          int h = col >> 6,  d = col & (HDq - 1);
          reinterpret_cast<unsigned short*>(out)
              [((size_t)(b * Hq + h) * Sq + s) * HDq + d] = f2bf_bits(v);
        }
      }
    }
  }
}

// ---------- Flash attention, transposed-scores formulation ----------
// S^T = K*Q^T: the exp'd score C-fragment (lane = q, elements = key) is exactly
// the A-fragment layout for P*V -> P never leaves registers.
__global__ __launch_bounds__(128) void attn_wmma(
    const unsigned short* __restrict__ Qh, const unsigned short* __restrict__ Kh,
    const unsigned short* __restrict__ Vh, const float* __restrict__ mask,
    unsigned short* __restrict__ ctx) {
  __shared__ unsigned short Ks[64 * 64];   // [key][d]
  __shared__ unsigned short VsT[64 * 64];  // [d][key] (transposed V tile)
  __shared__ float penS[64];

  const int nqt   = Sq / 64;
  const int blk   = blockIdx.x;
  const int qtile = blk % nqt;
  const int bh    = blk / nqt;
  const int h     = bh % Hq;
  const int b     = bh / Hq;

  const unsigned short* Qp = Qh + (size_t)bh * Sq * HDq;
  const unsigned short* Kp = Kh + (size_t)bh * Sq * HDq;
  const unsigned short* Vp = Vh + (size_t)bh * Sq * HDq;

  const int tid = threadIdx.x, wave = tid >> 5, lane = tid & 31;
  const int nloc = lane & 15, halfsel = lane >> 4;

  // Q as B-fragments (B[k=d][n=q]); row-major Q [q][d] IS the transposed-B layout.
  const int qrow0 = qtile * 64 + wave * 16;
  bf16x16 qb0 = load_frag_b_fastT(Qp + (size_t)qrow0 * HDq + 0,  HDq);
  bf16x16 qb1 = load_frag_b_fastT(Qp + (size_t)qrow0 * HDq + 32, HDq);

  floatx8 o[4] = {};
  float mst = -3.0e38f, lst = 0.0f;   // per-lane softmax stats for q = lane&15

  for (int j = 0; j < Sq; j += 64) {
    __syncthreads();
    {   // stage K block [key][d]: async copy (4x b128 per thread)
      int row = tid >> 1, colb = (tid & 1) * 32;
      const unsigned short* g = Kp + (size_t)(j + row) * HDq + colb;
      unsigned short* l = &Ks[row * 64 + colb];
#if HAVE_ASYNC_LDS
#pragma unroll
      for (int i = 0; i < 4; ++i) async_copy_b128(g + i * 8, l + i * 8);
#else
      const uint4* gq = reinterpret_cast<const uint4*>(g);
      uint4* lq = reinterpret_cast<uint4*>(l);
      lq[0] = gq[0]; lq[1] = gq[1]; lq[2] = gq[2]; lq[3] = gq[3];
#endif
    }
    {   // stage V block transposed -> VsT[d][key] with packed b32 writes
      int kp = tid & 31;          // key pair: keys 2kp, 2kp+1
      int nb = (tid >> 5) * 16;   // 16 d values
      const unsigned short* g0 = Vp + (size_t)(j + 2 * kp) * HDq + nb;
      unsigned int* VsT32 = reinterpret_cast<unsigned int*>(VsT);
#pragma unroll
      for (int g = 0; g < 2; ++g) {
        uint4 r0 = *reinterpret_cast<const uint4*>(g0 + g * 8);
        uint4 r1 = *reinterpret_cast<const uint4*>(g0 + HDq + g * 8);
        unsigned int w0[4] = {r0.x, r0.y, r0.z, r0.w};
        unsigned int w1[4] = {r1.x, r1.y, r1.z, r1.w};
#pragma unroll
        for (int i = 0; i < 8; ++i) {
          unsigned int lo = (i & 1) ? (w0[i >> 1] >> 16) : (w0[i >> 1] & 0xffffu);
          unsigned int hi = (i & 1) ? (w1[i >> 1] >> 16) : (w1[i >> 1] & 0xffffu);
          VsT32[(nb + g * 8 + i) * 32 + kp] = lo | (hi << 16);
        }
      }
    }
    if (tid < 64) penS[tid] = -1.0e6f * (1.0f - mask[(size_t)b * Sq + j + tid]);
    WAIT_ASYNC(0);
    __syncthreads();

    // S^T tile (64 key x 16 q): 4 key-subtiles, accumulated over d in 2 chunks
    floatx8 sc[4];
#pragma unroll
    for (int ms = 0; ms < 4; ++ms) {
      floatx8 z = {};
      bf16x16 ka0 = load_frag_a_fast(&Ks[(ms * 16) * 64 + 0], 64);
      z = __builtin_amdgcn_wmma_f32_16x16x32_bf16(false, ka0, false, qb0, (short)0, z, false, false);
      bf16x16 ka1 = load_frag_a_fast(&Ks[(ms * 16) * 64 + 32], 64);
      z = __builtin_amdgcn_wmma_f32_16x16x32_bf16(false, ka1, false, qb1, (short)0, z, false, false);
      sc[ms] = z;
    }

    // additive mask penalty: key = ms*16 + halfsel*8 + r (uniform across q lanes)
    const float4* pen4 = reinterpret_cast<const float4*>(penS);
#pragma unroll
    for (int ms = 0; ms < 4; ++ms) {
      int base4 = (ms * 16 + halfsel * 8) >> 2;
      float4 pA = pen4[base4], pB = pen4[base4 + 1];
      sc[ms][0] += pA.x; sc[ms][1] += pA.y; sc[ms][2] += pA.z; sc[ms][3] += pA.w;
      sc[ms][4] += pB.x; sc[ms][5] += pB.y; sc[ms][6] += pB.z; sc[ms][7] += pB.w;
    }

    // online softmax over keys: per-lane partial + one xor-16 combine
    float lmax = sc[0][0];
#pragma unroll
    for (int ms = 0; ms < 4; ++ms)
#pragma unroll
      for (int r = 0; r < 8; ++r) lmax = fmaxf(lmax, sc[ms][r]);
    lmax = fmaxf(lmax, __shfl_xor(lmax, 16, 32));
    float mnew  = fmaxf(mst, lmax);
    float alpha = __expf(mst - mnew);
    mst = mnew;

    float rs = 0.0f;
#pragma unroll
    for (int ms = 0; ms < 4; ++ms)
#pragma unroll
      for (int r = 0; r < 8; ++r) {
        float p = __expf(sc[ms][r] - mnew);
        sc[ms][r] = p;
        rs += p;
      }
    rs += __shfl_xor(rs, 16, 32);
    lst = lst * alpha + rs;

    // rescale O: element r corresponds to q-row halfsel*8 + r -> gather alpha
    float ar[8];
#pragma unroll
    for (int r = 0; r < 8; ++r) ar[r] = __shfl(alpha, halfsel * 8 + r, 32);
#pragma unroll
    for (int ns = 0; ns < 4; ++ns)
#pragma unroll
      for (int r = 0; r < 8; ++r) o[ns][r] *= ar[r];

    // P fragments: score C-layout == A-fragment layout (registers only)
    bf16x16 pa[2];
#pragma unroll
    for (int c = 0; c < 2; ++c) {
#pragma unroll
      for (int e = 0; e < 8; ++e) pa[c][e]     = bits2bf(f2bf_bits(sc[2 * c][e]));
#pragma unroll
      for (int e = 0; e < 8; ++e) pa[c][8 + e] = bits2bf(f2bf_bits(sc[2 * c + 1][e]));
    }

    // O += P * V   (B fragments from transposed VsT)
#pragma unroll
    for (int ns = 0; ns < 4; ++ns) {
      bf16x16 vb0 = load_frag_b_fastT(&VsT[(ns * 16) * 64 + 0], 64);
      o[ns] = __builtin_amdgcn_wmma_f32_16x16x32_bf16(false, pa[0], false, vb0, (short)0, o[ns], false, false);
      bf16x16 vb1 = load_frag_b_fastT(&VsT[(ns * 16) * 64 + 32], 64);
      o[ns] = __builtin_amdgcn_wmma_f32_16x16x32_bf16(false, pa[1], false, vb1, (short)0, o[ns], false, false);
    }
  }

  // normalize and store ctx bf16 [B, S, H*HD]
  float lr[8];
#pragma unroll
  for (int r = 0; r < 8; ++r) {
    float lv = __shfl(lst, halfsel * 8 + r, 32);
    lr[r] = 1.0f / fmaxf(lv, 1e-30f);
  }
#pragma unroll
  for (int ns = 0; ns < 4; ++ns)
#pragma unroll
    for (int r = 0; r < 8; ++r) {
      int srow = qtile * 64 + wave * 16 + halfsel * 8 + r;
      int col  = h * HDq + ns * 16 + nloc;
      ctx[((size_t)b * Sq + srow) * (Hq * HDq) + col] = f2bf_bits(o[ns][r] * lr[r]);
    }
}

// ---------- host-side launch ----------
extern "C" void kernel_launch(void* const* d_in, const int* in_sizes, int n_in,
                              void* d_out, int out_size, void* d_ws, size_t ws_size,
                              hipStream_t stream) {
  const float* X    = (const float*)d_in[0];
  const float* mask = (const float*)d_in[1];
  const float* Wq   = (const float*)d_in[2];
  const float* bq   = (const float*)d_in[3];
  const float* Wk   = (const float*)d_in[4];
  const float* bk   = (const float*)d_in[5];
  const float* Wv   = (const float*)d_in[6];
  const float* bv   = (const float*)d_in[7];
  const float* Wo   = (const float*)d_in[8];
  const float* bo   = (const float*)d_in[9];
  float* out = (float*)d_out;

  const size_t M = (size_t)Bq * Sq;   // 8192
  const size_t N = (size_t)Dq;        // 1024 == H*HD

  char* ws = (char*)d_ws;
  unsigned short* Xb  = (unsigned short*)ws; ws += M * N * 2;
  unsigned short* Wqb = (unsigned short*)ws; ws += N * N * 2;
  unsigned short* Wkb = (unsigned short*)ws; ws += N * N * 2;
  unsigned short* Wvb = (unsigned short*)ws; ws += N * N * 2;
  unsigned short* Wob = (unsigned short*)ws; ws += N * N * 2;
  unsigned short* Qh  = (unsigned short*)ws; ws += M * N * 2;
  unsigned short* Kh  = (unsigned short*)ws; ws += M * N * 2;
  unsigned short* Vh  = (unsigned short*)ws; ws += M * N * 2;
  unsigned short* Ctx = (unsigned short*)ws; ws += M * N * 2;

  auto cvt = [&](const float* src, unsigned short* dst, size_t n) {
    unsigned g = (unsigned)((n + 255) / 256);
    f32_to_bf16_kernel<<<g, 256, 0, stream>>>(src, dst, n);
  };
  cvt(X,  Xb,  M * N);
  cvt(Wq, Wqb, N * N);
  cvt(Wk, Wkb, N * N);
  cvt(Wv, Wvb, N * N);
  cvt(Wo, Wob, N * N);

  dim3 gg((unsigned)(M / 128), (unsigned)(N / 64));
  // Q pre-scaled by 1/sqrt(HD)=0.125 (folds attention scale into Q)
  gemm_bf16_wmma<1><<<gg, 128, 0, stream>>>(Xb, Wqb, bq, Qh, (int)N, Dq, 0.125f);
  gemm_bf16_wmma<1><<<gg, 128, 0, stream>>>(Xb, Wkb, bk, Kh, (int)N, Dq, 1.0f);
  gemm_bf16_wmma<1><<<gg, 128, 0, stream>>>(Xb, Wvb, bv, Vh, (int)N, Dq, 1.0f);

  attn_wmma<<<(unsigned)(Bq * Hq * (Sq / 64)), 128, 0, stream>>>(Qh, Kh, Vh, mask, Ctx);

  gemm_bf16_wmma<0><<<gg, 128, 0, stream>>>(Ctx, Wob, bo, out, (int)N, (int)N, 1.0f);
}
